// PointNet2_54168127537474
// MI455X (gfx1250) — matrix operand order repaired and checked by reading, not compiled, not run on hardware
//
#include <hip/hip_runtime.h>

typedef __attribute__((ext_vector_type(16))) _Float16 v16h;
typedef __attribute__((ext_vector_type(8)))  _Float16 v8h;
typedef __attribute__((ext_vector_type(8)))  float    v8f;

// ---------------------------------------------------------------- utilities
__global__ void k_split(const float* __restrict__ xyz8, float* __restrict__ l0_xyz,
                        float* __restrict__ l0_pts, int N) {
  int i = blockIdx.x * blockDim.x + threadIdx.x;
  if (i >= N) return;
  l0_xyz[i*3+0] = xyz8[i*8+0];
  l0_xyz[i*3+1] = xyz8[i*8+1];
  l0_xyz[i*3+2] = xyz8[i*8+2];
  for (int d = 0; d < 5; ++d) l0_pts[i*5+d] = xyz8[i*8+3+d];
}

// Furthest point sampling: single persistent block, LDS argmax tree.
__global__ void __launch_bounds__(1024) k_fps(const float* __restrict__ xyz, int N,
                                              int npoint, int* __restrict__ idx_out,
                                              float* __restrict__ dist) {
  const int T = blockDim.x;
  const int t = threadIdx.x;
  __shared__ float s_val[1024];
  __shared__ int   s_idx[1024];
  __shared__ float s_c[3];
  for (int i = t; i < N; i += T) dist[i] = 1e10f;
  int far = 0;
  __syncthreads();
  for (int it = 0; it < npoint; ++it) {
    if (t == 0) {
      idx_out[it] = far;
      s_c[0] = xyz[far*3+0]; s_c[1] = xyz[far*3+1]; s_c[2] = xyz[far*3+2];
    }
    __syncthreads();
    float cx = s_c[0], cy = s_c[1], cz = s_c[2];
    float best = -1.0f; int bi = N;
    for (int i = t; i < N; i += T) {
      float dx = xyz[i*3+0]-cx, dy = xyz[i*3+1]-cy, dz = xyz[i*3+2]-cz;
      float d = dx*dx + dy*dy + dz*dz;
      float dm = dist[i];
      if (d < dm) { dm = d; dist[i] = dm; }
      if (dm > best) { best = dm; bi = i; }   // keeps lowest i per thread
    }
    s_val[t] = best; s_idx[t] = bi;
    __syncthreads();
    for (int s = T >> 1; s > 0; s >>= 1) {
      if (t < s) {
        float v2 = s_val[t+s]; int i2 = s_idx[t+s];
        if (v2 > s_val[t] || (v2 == s_val[t] && i2 < s_idx[t])) { s_val[t] = v2; s_idx[t] = i2; }
      }
      __syncthreads();
    }
    far = s_idx[0];
    __syncthreads();
  }
}

__global__ void k_gather_xyz(const float* __restrict__ xyz, const int* __restrict__ idx,
                             float* __restrict__ out, int n) {
  int i = blockIdx.x * blockDim.x + threadIdx.x;
  if (i >= n) return;
  int j = idx[i];
  out[i*3+0] = xyz[j*3+0]; out[i*3+1] = xyz[j*3+1]; out[i*3+2] = xyz[j*3+2];
}

// Ball query with CUDA semantics: first nsample in index order; fill with first (or 0).
__global__ void k_ballquery(const float* __restrict__ xyz, int N,
                            const float* __restrict__ new_xyz, int npoint,
                            float r2, int nsample, int* __restrict__ out) {
  int p = blockIdx.x * blockDim.x + threadIdx.x;
  if (p >= npoint) return;
  float cx = new_xyz[p*3+0], cy = new_xyz[p*3+1], cz = new_xyz[p*3+2];
  int cnt = 0, first = 0;
  bool found = false;
  int* o = out + (size_t)p * nsample;
  for (int j = 0; j < N && cnt < nsample; ++j) {
    float dx = xyz[j*3+0]-cx, dy = xyz[j*3+1]-cy, dz = xyz[j*3+2]-cz;
    if (dx*dx + dy*dy + dz*dz <= r2) {
      if (!found) { found = true; first = j; }
      o[cnt++] = j;
    }
  }
  for (int s = cnt; s < nsample; ++s) o[s] = first;
}

// Build grouped A matrix [npoint*nsample, Kpad] f16: [xyz-center | feats | 0-pad]
__global__ void k_group(const float* __restrict__ xyz, const float* __restrict__ pts, int C,
                        const float* __restrict__ new_xyz, const int* __restrict__ bq,
                        int nsample, long total, int Kpad, _Float16* __restrict__ A) {
  long t = (long)blockIdx.x * blockDim.x + threadIdx.x;
  if (t >= total) return;
  int c = (int)(t % Kpad);
  long r = t / Kpad;
  int p = (int)(r / nsample);
  int j = bq[r];
  float v = 0.0f;
  if (c < 3)          v = xyz[(size_t)j*3 + c] - new_xyz[(size_t)p*3 + c];
  else if (c < 3 + C) v = pts[(size_t)j*C + (c-3)];
  A[t] = (_Float16)v;
}

// Transpose + pad weights: W[cin,cout] f32 -> Bt[cout,Kpad] f16 (zero pad K>=cin)
__global__ void k_prep_w(const float* __restrict__ W, int cin, int cout, int Kpad,
                         _Float16* __restrict__ Bt) {
  int t = blockIdx.x * blockDim.x + threadIdx.x;
  if (t >= cout * Kpad) return;
  int k = t % Kpad, o = t / Kpad;
  Bt[t] = (_Float16)((k < cin) ? W[(size_t)k*cout + o] : 0.0f);
}

// WMMA GEMM: C[rows,cout] = A[rows,Kpad] * Bt[cout,Kpad]^T
// One 32x64 macro-tile per wave: 2 M-frags x 4 N-frags -> 8 accumulators.
// 12 b128 fragment loads per 8 WMMAs (vs 4 loads/WMMA for a single tile).
__global__ void __launch_bounds__(256) k_gemm_wmma(const _Float16* __restrict__ A,
                                                   const _Float16* __restrict__ Bt,
                                                   _Float16* __restrict__ C,
                                                   int Mtiles, int Ntiles, int Kpad, int ldc) {
  // Mtiles = rows/32, Ntiles = cout/64
  int wave = threadIdx.x >> 5;
  int lane = threadIdx.x & 31;
  int tile = blockIdx.x * (blockDim.x >> 5) + wave;
  if (tile >= Mtiles * Ntiles) return;
  int nt = tile % Ntiles, mt = tile / Ntiles;
  int g = lane >> 4;        // lane half-group
  int l = lane & 15;
  // A frag rows: mi*16 + l ; halves 0..15 -> K = k0 + 16*g + h (contiguous 16)
  const _Float16* arow0 = A + (size_t)(mt*32 + l) * Kpad + 16*g;
  const _Float16* arow1 = arow0 + (size_t)16 * Kpad;
  // B frag cols: nt*64 + j*16 + l ; halves 0..15 -> K = k0 + 16*g + h (contiguous 16)
  const _Float16* brow  = Bt + (size_t)(nt*64 + l) * Kpad + 16*g;
  const size_t bstride = (size_t)16 * Kpad;

  v8f acc[2][4];
#pragma unroll
  for (int mi = 0; mi < 2; ++mi)
#pragma unroll
    for (int j = 0; j < 4; ++j) acc[mi][j] = (v8f){};

  for (int k0 = 0; k0 < Kpad; k0 += 32) {
    v16h a[2];
    {
      v8h lo0 = *(const v8h*)(arow0 + k0);
      v8h hi0 = *(const v8h*)(arow0 + k0 + 8);
      v8h lo1 = *(const v8h*)(arow1 + k0);
      v8h hi1 = *(const v8h*)(arow1 + k0 + 8);
#pragma unroll
      for (int i = 0; i < 8; ++i) {
        a[0][i] = lo0[i]; a[0][i+8] = hi0[i];
        a[1][i] = lo1[i]; a[1][i+8] = hi1[i];
      }
    }
    v16h b[4];
#pragma unroll
    for (int j = 0; j < 4; ++j) b[j] = *(const v16h*)(brow + bstride*j + k0);
#pragma unroll
    for (int mi = 0; mi < 2; ++mi)
#pragma unroll
      for (int j = 0; j < 4; ++j)
        acc[mi][j] = __builtin_amdgcn_wmma_f32_16x16x32_f16(false, a[mi], false, b[j],
                                                            (short)0, acc[mi][j], false, false);
  }

#pragma unroll
  for (int mi = 0; mi < 2; ++mi) {
#pragma unroll
    for (int j = 0; j < 4; ++j) {
#pragma unroll
      for (int r = 0; r < 8; ++r) {
        int row = mt*32 + mi*16 + r + 8*g;
        C[(size_t)row * ldc + nt*64 + j*16 + l] = (_Float16)acc[mi][j][r];
      }
    }
  }
}

// Per-channel BN stats -> affine (a,b): y = a*x + b  (bias cancels under BN)
__global__ void k_bnstats(const _Float16* __restrict__ C, long rows, int ldc,
                          const float* __restrict__ gamma, const float* __restrict__ beta,
                          float* __restrict__ ba, float* __restrict__ bb) {
  int c = blockIdx.x;
  int t = threadIdx.x;
  const int T = blockDim.x;
  float s = 0.0f, s2 = 0.0f;
  for (long r = t; r < rows; r += T) {
    float x = (float)C[r * ldc + c];
    s += x; s2 += x * x;
  }
  __shared__ float ss[256], sq[256];
  ss[t] = s; sq[t] = s2;
  __syncthreads();
  for (int k = T >> 1; k > 0; k >>= 1) {
    if (t < k) { ss[t] += ss[t+k]; sq[t] += sq[t+k]; }
    __syncthreads();
  }
  if (t == 0) {
    float inv = 1.0f / (float)rows;
    float mu  = ss[0] * inv;
    float var = sq[0] * inv - mu * mu;
    float sc  = gamma[c] * rsqrtf(var + 1e-5f);
    ba[c] = sc;
    bb[c] = beta[c] - mu * sc;
  }
}

__global__ void k_bnrelu(_Float16* __restrict__ C, long total, int ldc, int cout,
                         const float* __restrict__ ba, const float* __restrict__ bb) {
  long t = (long)blockIdx.x * blockDim.x + threadIdx.x;
  if (t >= total) return;
  int c = (int)(t % ldc);
  float y = 0.0f;
  if (c < cout) {
    float x = (float)C[t];
    y = fmaxf(0.0f, x * ba[c] + bb[c]);
  }
  C[t] = (_Float16)y;
}

__global__ void k_maxpool(const _Float16* __restrict__ C, int npoint, int nsample, int cout,
                          float* __restrict__ out) {
  int t = blockIdx.x * blockDim.x + threadIdx.x;
  if (t >= npoint * cout) return;
  int c = t % cout, p = t / cout;
  const _Float16* base = C + (size_t)p * nsample * cout + c;
  float m = -1e30f;
  for (int s = 0; s < nsample; ++s) m = fmaxf(m, (float)base[(size_t)s * cout]);
  out[t] = m;
}

// 3-NN (first-occurrence tie order) + inverse-distance weights
__global__ void k_knn3(const float* __restrict__ uxyz, int nu,
                       const float* __restrict__ kxyz, int nk,
                       int* __restrict__ idx3, float* __restrict__ w3) {
  int i = blockIdx.x * blockDim.x + threadIdx.x;
  if (i >= nu) return;
  float cx = uxyz[i*3+0], cy = uxyz[i*3+1], cz = uxyz[i*3+2];
  float d0 = 1e30f, d1 = 1e30f, d2 = 1e30f;
  int i0 = 0, i1 = 0, i2 = 0;
  for (int j = 0; j < nk; ++j) {
    float dx = kxyz[j*3+0]-cx, dy = kxyz[j*3+1]-cy, dz = kxyz[j*3+2]-cz;
    float d = dx*dx + dy*dy + dz*dz;
    if (d < d0)      { d2=d1; i2=i1; d1=d0; i1=i0; d0=d; i0=j; }
    else if (d < d1) { d2=d1; i2=i1; d1=d;  i1=j; }
    else if (d < d2) { d2=d;  i2=j; }
  }
  float w0 = 1.0f / (sqrtf(fmaxf(d0, 0.0f)) + 1e-8f);
  float w1 = 1.0f / (sqrtf(fmaxf(d1, 0.0f)) + 1e-8f);
  float w2 = 1.0f / (sqrtf(fmaxf(d2, 0.0f)) + 1e-8f);
  float s = w0 + w1 + w2;
  idx3[i*3+0] = i0; idx3[i*3+1] = i1; idx3[i*3+2] = i2;
  w3[i*3+0] = w0/s; w3[i*3+1] = w1/s; w3[i*3+2] = w2/s;
}

// Build FP input A = [unknown_feats(C1) | interp(C2) | 0-pad] as f16 [nu, Kpad]
__global__ void k_fp_build(const float* __restrict__ upts, int C1,
                           const float* __restrict__ kpts, int C2,
                           const int* __restrict__ idx3, const float* __restrict__ w3,
                           long total, int Kpad, _Float16* __restrict__ A) {
  long t = (long)blockIdx.x * blockDim.x + threadIdx.x;
  if (t >= total) return;
  int c = (int)(t % Kpad);
  int i = (int)(t / Kpad);
  float v = 0.0f;
  if (c < C1) v = upts[(size_t)i*C1 + c];
  else if (c < C1 + C2) {
    int cc = c - C1;
    v = w3[i*3+0] * kpts[(size_t)idx3[i*3+0]*C2 + cc]
      + w3[i*3+1] * kpts[(size_t)idx3[i*3+1]*C2 + cc]
      + w3[i*3+2] * kpts[(size_t)idx3[i*3+2]*C2 + cc];
  }
  A[t] = (_Float16)v;
}

__global__ void k_h2f(const _Float16* __restrict__ in, float* __restrict__ out, long n) {
  long t = (long)blockIdx.x * blockDim.x + threadIdx.x;
  if (t < n) out[t] = (float)in[t];
}

// Final tiny head: out[i,c] = b2[c] + sum_k X[i,k]*W2[k,c]  (f32 out)
__global__ void k_head2(const _Float16* __restrict__ X, const float* __restrict__ W,
                        const float* __restrict__ b, float* __restrict__ out,
                        int n, int K, int Co) {
  int t = blockIdx.x * blockDim.x + threadIdx.x;
  if (t >= n * Co) return;
  int c = t % Co, i = t / Co;
  float s = b[c];
  for (int k = 0; k < K; ++k) s += (float)X[(size_t)i*K + k] * W[(size_t)k*Co + c];
  out[t] = s;
}

// ---------------------------------------------------------------- host side
namespace {

struct Bump {
  char* base; size_t off;
  void* take(size_t bytes) {
    off = (off + 255) & ~(size_t)255;
    void* r = base + off;
    off += bytes;
    return r;
  }
};

inline int pad32(int x) { return (x + 31) & ~31; }

void run_layer(hipStream_t st, const _Float16* A, long rows, int Kpad,
               const float* W, int cin, int cout,
               const float* gamma, const float* beta,
               _Float16* wbuf, float* bn_a, float* bn_b, _Float16* Cout) {
  int wp = cout * Kpad;
  k_prep_w<<<(wp + 255) / 256, 256, 0, st>>>(W, cin, cout, Kpad, wbuf);
  int Mtiles = (int)(rows / 32);     // 32-row macro tiles
  int Ntiles = cout / 64;            // 64-col macro tiles (all couts are multiples of 64)
  long tiles = (long)Mtiles * Ntiles;
  k_gemm_wmma<<<(int)((tiles + 7) / 8), 256, 0, st>>>(A, wbuf, Cout, Mtiles, Ntiles, Kpad, cout);
  k_bnstats<<<cout, 256, 0, st>>>(Cout, rows, cout, gamma, beta, bn_a, bn_b);
  long tot = rows * (long)cout;
  k_bnrelu<<<(int)((tot + 255) / 256), 256, 0, st>>>(Cout, tot, cout, cout, bn_a, bn_b);
}

} // namespace

extern "C" void kernel_launch(void* const* d_in, const int* in_sizes, int n_in,
                              void* d_out, int out_size, void* d_ws, size_t ws_size,
                              hipStream_t stream) {
  (void)in_sizes; (void)n_in; (void)out_size; (void)ws_size;
  auto F = [&](int i) { return (const float*)d_in[i]; };

  const int N = 16384;
  const float* xyz8 = F(0);
  // param layout (insertion order): sa1(3x4) sa2(3x4) sa3(3x4) fp3(2x4) fp2(2x4) fp1(2x4) head1(4) head2(2)
  const int SA1 = 1, SA2 = 13, SA3 = 25, FP3 = 37, FP2 = 45, FP1 = 53, H1 = 61, H2 = 65;

  Bump ws{(char*)d_ws, 0};
  float* l0_xyz  = (float*)ws.take((size_t)N*3*4);
  float* l0_pts  = (float*)ws.take((size_t)N*5*4);
  float* dist    = (float*)ws.take((size_t)N*4);
  int*   fps_idx = (int*)  ws.take((size_t)8192*4);
  int*   bq      = (int*)  ws.take((size_t)262144*4);
  float* l1_xyz  = (float*)ws.take((size_t)8192*3*4);
  float* l2_xyz  = (float*)ws.take((size_t)4096*3*4);
  float* l3_xyz  = (float*)ws.take((size_t)2048*3*4);
  float* l1_pts  = (float*)ws.take((size_t)8192*128*4);
  float* l2_pts  = (float*)ws.take((size_t)4096*256*4);
  float* l3_pts  = (float*)ws.take((size_t)2048*256*4);
  float* l1_ptsB = (float*)ws.take((size_t)8192*256*4);
  int*   idx3    = (int*)  ws.take((size_t)N*3*4);
  float* w3      = (float*)ws.take((size_t)N*3*4);
  float* bn_a    = (float*)ws.take(256*4);
  float* bn_b    = (float*)ws.take(256*4);
  _Float16* wbuf = (_Float16*)ws.take((size_t)512*512*2);
  const size_t BUF_HALVES = (size_t)262144 * 256;    // largest activation tensor (SA2 out)
  _Float16* bufA = (_Float16*)ws.take(BUF_HALVES*2);
  _Float16* bufB = (_Float16*)ws.take(BUF_HALVES*2);

  k_split<<<(N + 255) / 256, 256, 0, stream>>>(xyz8, l0_xyz, l0_pts, N);

  // ----- SA1: 16384 -> 8192 centers, r=1, ns=16, cin=8 (Kpad 32) -> 64 -> 64 -> 128
  {
    k_fps<<<1, 1024, 0, stream>>>(l0_xyz, N, 8192, fps_idx, dist);
    k_gather_xyz<<<(8192 + 255) / 256, 256, 0, stream>>>(l0_xyz, fps_idx, l1_xyz, 8192);
    k_ballquery<<<(8192 + 255) / 256, 256, 0, stream>>>(l0_xyz, N, l1_xyz, 8192, 1.0f, 16, bq);
    long rows = (long)8192 * 16;
    int Kp = pad32(8);
    long tot = rows * Kp;
    k_group<<<(int)((tot + 255) / 256), 256, 0, stream>>>(l0_xyz, l0_pts, 5, l1_xyz, bq, 16, tot, Kp, bufA);
    run_layer(stream, bufA, rows, Kp,  F(SA1+0), 8,  64,  F(SA1+2),  F(SA1+3),  wbuf, bn_a, bn_b, bufB);
    run_layer(stream, bufB, rows, 64,  F(SA1+4), 64, 64,  F(SA1+6),  F(SA1+7),  wbuf, bn_a, bn_b, bufA);
    run_layer(stream, bufA, rows, 64,  F(SA1+8), 64, 128, F(SA1+10), F(SA1+11), wbuf, bn_a, bn_b, bufB);
    k_maxpool<<<(8192*128 + 255) / 256, 256, 0, stream>>>(bufB, 8192, 16, 128, l1_pts);
  }

  // ----- SA2: 8192 -> 4096, r=5, ns=64, cin=131 (Kpad 160) -> 128 -> 128 -> 256
  {
    k_fps<<<1, 1024, 0, stream>>>(l1_xyz, 8192, 4096, fps_idx, dist);
    k_gather_xyz<<<(4096 + 255) / 256, 256, 0, stream>>>(l1_xyz, fps_idx, l2_xyz, 4096);
    k_ballquery<<<(4096 + 255) / 256, 256, 0, stream>>>(l1_xyz, 8192, l2_xyz, 4096, 25.0f, 64, bq);
    long rows = (long)4096 * 64;
    int Kp = pad32(131);
    long tot = rows * Kp;
    k_group<<<(int)((tot + 255) / 256), 256, 0, stream>>>(l1_xyz, l1_pts, 128, l2_xyz, bq, 64, tot, Kp, bufA);
    run_layer(stream, bufA, rows, Kp,  F(SA2+0), 131, 128, F(SA2+2),  F(SA2+3),  wbuf, bn_a, bn_b, bufB);
    run_layer(stream, bufB, rows, 128, F(SA2+4), 128, 128, F(SA2+6),  F(SA2+7),  wbuf, bn_a, bn_b, bufA);
    run_layer(stream, bufA, rows, 128, F(SA2+8), 128, 256, F(SA2+10), F(SA2+11), wbuf, bn_a, bn_b, bufB);
    k_maxpool<<<(4096*256 + 255) / 256, 256, 0, stream>>>(bufB, 4096, 64, 256, l2_pts);
  }

  // ----- SA3: 4096 -> 2048, r=15, ns=64, cin=259 (Kpad 288) -> 128 -> 128 -> 256
  {
    k_fps<<<1, 1024, 0, stream>>>(l2_xyz, 4096, 2048, fps_idx, dist);
    k_gather_xyz<<<(2048 + 255) / 256, 256, 0, stream>>>(l2_xyz, fps_idx, l3_xyz, 2048);
    k_ballquery<<<(2048 + 255) / 256, 256, 0, stream>>>(l2_xyz, 4096, l3_xyz, 2048, 225.0f, 64, bq);
    long rows = (long)2048 * 64;
    int Kp = pad32(259);
    long tot = rows * Kp;
    k_group<<<(int)((tot + 255) / 256), 256, 0, stream>>>(l2_xyz, l2_pts, 256, l3_xyz, bq, 64, tot, Kp, bufA);
    run_layer(stream, bufA, rows, Kp,  F(SA3+0), 259, 128, F(SA3+2),  F(SA3+3),  wbuf, bn_a, bn_b, bufB);
    run_layer(stream, bufB, rows, 128, F(SA3+4), 128, 128, F(SA3+6),  F(SA3+7),  wbuf, bn_a, bn_b, bufA);
    run_layer(stream, bufA, rows, 128, F(SA3+8), 128, 256, F(SA3+10), F(SA3+11), wbuf, bn_a, bn_b, bufB);
    k_maxpool<<<(2048*256 + 255) / 256, 256, 0, stream>>>(bufB, 2048, 64, 256, l3_pts);
  }

  // ----- FP3: unknown=l2 (4096), known=l3 (2048), cin 512 -> 256 -> 256
  {
    int nu = 4096;
    k_knn3<<<(nu + 255) / 256, 256, 0, stream>>>(l2_xyz, nu, l3_xyz, 2048, idx3, w3);
    int Kp = 512;
    long tot = (long)nu * Kp;
    k_fp_build<<<(int)((tot + 255) / 256), 256, 0, stream>>>(l2_pts, 256, l3_pts, 256, idx3, w3, tot, Kp, bufA);
    run_layer(stream, bufA, nu, Kp,  F(FP3+0), 512, 256, F(FP3+2), F(FP3+3), wbuf, bn_a, bn_b, bufB);
    run_layer(stream, bufB, nu, 256, F(FP3+4), 256, 256, F(FP3+6), F(FP3+7), wbuf, bn_a, bn_b, bufA);
    k_h2f<<<(int)(((long)nu*256 + 255) / 256), 256, 0, stream>>>(bufA, l2_pts, (long)nu*256);
  }

  // ----- FP2: unknown=l1 (8192), known=l2 (4096), cin 384 -> 256 -> 256
  {
    int nu = 8192;
    k_knn3<<<(nu + 255) / 256, 256, 0, stream>>>(l1_xyz, nu, l2_xyz, 4096, idx3, w3);
    int Kp = 384;
    long tot = (long)nu * Kp;
    k_fp_build<<<(int)((tot + 255) / 256), 256, 0, stream>>>(l1_pts, 128, l2_pts, 256, idx3, w3, tot, Kp, bufA);
    run_layer(stream, bufA, nu, Kp,  F(FP2+0), 384, 256, F(FP2+2), F(FP2+3), wbuf, bn_a, bn_b, bufB);
    run_layer(stream, bufB, nu, 256, F(FP2+4), 256, 256, F(FP2+6), F(FP2+7), wbuf, bn_a, bn_b, bufA);
    k_h2f<<<(int)(((long)nu*256 + 255) / 256), 256, 0, stream>>>(bufA, l1_ptsB, (long)nu*256);
  }

  // ----- FP1: unknown=l0 (16384), known=l1, cin 261 (Kpad 288) -> 128 -> 64
  {
    int nu = N;
    k_knn3<<<(nu + 255) / 256, 256, 0, stream>>>(l0_xyz, nu, l1_xyz, 8192, idx3, w3);
    int Kp = pad32(261);
    long tot = (long)nu * Kp;
    k_fp_build<<<(int)((tot + 255) / 256), 256, 0, stream>>>(l0_pts, 5, l1_ptsB, 256, idx3, w3, tot, Kp, bufA);
    run_layer(stream, bufA, nu, Kp,  F(FP1+0), 261, 128, F(FP1+2), F(FP1+3), wbuf, bn_a, bn_b, bufB);
    run_layer(stream, bufB, nu, 128, F(FP1+4), 128, 64,  F(FP1+6), F(FP1+7), wbuf, bn_a, bn_b, bufA);
  }

  // ----- head1 (WMMA + BN + ReLU), head2 (tiny VALU GEMM with bias, f32 out)
  run_layer(stream, bufA, N, 64, F(H1+0), 64, 64, F(H1+2), F(H1+3), wbuf, bn_a, bn_b, bufB);
  k_head2<<<(N*8 + 255) / 256, 256, 0, stream>>>(bufB, F(H2+0), F(H2+1), (float*)d_out, N, 64, 8);
}
